// MambaBlock_85298050499090
// MI455X (gfx1250) — matrix operand order repaired and checked by reading, not compiled, not run on hardware
//
#include <hip/hip_runtime.h>
#include <hip/hip_bf16.h>

// ---------------------------------------------------------------------------
// Mamba block forward for gfx1250 (MI455X).
// GEMMs run on the WMMA path (v_wmma_f32_16x16x32_bf16, fp32 accumulate).
// Conv / softplus / selective scan / SiLU stay fp32 on the VALU.
// ---------------------------------------------------------------------------

#define BATCH    2
#define SEQ      1024
#define DMODEL   1024
#define DSTATE   16
#define DCONV    4
#define DINNER   2048              // EXPAND * DMODEL
#define ROWS     (BATCH * SEQ)     // 2048
#define NXZ      (2 * DINNER)      // 4096
#define NDBU     (2 * DSTATE + DINNER) // 2080

typedef __bf16 bf16_t;
typedef __attribute__((ext_vector_type(8)))  __bf16 v8bf;
typedef __attribute__((ext_vector_type(16))) __bf16 v16bf;
typedef __attribute__((ext_vector_type(8)))  float  v8f;

union FragBF {
    v16bf v;
    v8bf  h[2];
};

static __device__ __forceinline__ bf16_t f2bf(float f) {
    unsigned int u = __builtin_bit_cast(unsigned int, f);
    unsigned int r = u + 0x7FFFu + ((u >> 16) & 1u);   // round to nearest even
    unsigned short hs = (unsigned short)(r >> 16);
    return __builtin_bit_cast(bf16_t, hs);
}

// ---------------------------------------------------------------------------
// fp32 -> bf16 conversion
// ---------------------------------------------------------------------------
__global__ void cvt_f32_to_bf16(const float* __restrict__ in,
                                bf16_t* __restrict__ out, int n) {
    int i = blockIdx.x * blockDim.x + threadIdx.x;
    if (i < n) out[i] = f2bf(in[i]);
}

// A = -exp(A_log)
__global__ void neg_exp_kernel(const float* __restrict__ in,
                               float* __restrict__ out, int n) {
    int i = blockIdx.x * blockDim.x + threadIdx.x;
    if (i < n) out[i] = -__expf(in[i]);
}

// ---------------------------------------------------------------------------
// Generic GEMM: C[M,N] = A[M,K](bf16) * W[N,K]^T(bf16) + bias[N]  (fp32 out)
// 256 threads = 8 waves; waves tiled 4(M) x 2(N); each wave owns a 32x32
// output tile via 2x2 v_wmma_f32_16x16x32_bf16 fragments. K step = 32.
//
// A fragment (this code's contiguous K order, same as B): lane = M (lane&15), half-wave selects
// K 0..15 vs 16..31 -> two 16B loads per lane.
// B fragment (32x16, K-major per lane): lane = N, half-wave selects
// K 0..15 vs 16..31 -> one 32B contiguous load per lane.
// ---------------------------------------------------------------------------
__global__ __launch_bounds__(256, 2)
void gemm_bf16_wmma(const bf16_t* __restrict__ A,  // M x K row-major
                    const bf16_t* __restrict__ W,  // N x K row-major
                    const float*  __restrict__ bias,
                    float* __restrict__ C,
                    int M, int N, int K, int ldc) {
    const int lane = threadIdx.x & 31;
    const int wave = threadIdx.x >> 5;       // 0..7
    const int wm   = wave >> 1;              // 0..3
    const int wn   = wave & 1;               // 0..1
    const int m0   = blockIdx.y * 128 + wm * 32;
    const int n0   = blockIdx.x * 64  + wn * 32;
    if (n0 >= N || m0 >= M) return;

    const int l16 = lane & 15;
    const int h   = lane >> 4;

    const bf16_t* a0p = A + (size_t)(m0 + l16)      * K + h * 16;
    const bf16_t* a1p = A + (size_t)(m0 + 16 + l16) * K + h * 16;
    const bf16_t* b0p = W + (size_t)(n0 + l16)      * K + h * 16;
    const bf16_t* b1p = W + (size_t)(n0 + 16 + l16) * K + h * 16;

    v8f acc00 = {}, acc01 = {}, acc10 = {}, acc11 = {};

#pragma unroll 2
    for (int k = 0; k < K; k += 32) {
        if (k + 256 < K) {
            __builtin_prefetch(a0p + k + 256, 0, 1);
            __builtin_prefetch(b0p + k + 256, 0, 1);
        }
        FragBF a0, a1;
        a0.h[0] = *(const v8bf*)(a0p + k);
        a0.h[1] = *(const v8bf*)(a0p + k + 8);
        a1.h[0] = *(const v8bf*)(a1p + k);
        a1.h[1] = *(const v8bf*)(a1p + k + 8);
        v16bf b0 = *(const v16bf*)(b0p + k);
        v16bf b1 = *(const v16bf*)(b1p + k);

        acc00 = __builtin_amdgcn_wmma_f32_16x16x32_bf16(
            false, a0.v, false, b0, (short)0, acc00, false, false);
        acc01 = __builtin_amdgcn_wmma_f32_16x16x32_bf16(
            false, a0.v, false, b1, (short)0, acc01, false, false);
        acc10 = __builtin_amdgcn_wmma_f32_16x16x32_bf16(
            false, a1.v, false, b0, (short)0, acc10, false, false);
        acc11 = __builtin_amdgcn_wmma_f32_16x16x32_bf16(
            false, a1.v, false, b1, (short)0, acc11, false, false);
    }

    // C/D layout: VGPR r holds M = r (lanes 0-15) / r+8 (lanes 16-31),
    // lane&15 selects N within the 16-wide fragment.
    const int   c0 = n0 + l16;
    const int   c1 = n0 + 16 + l16;
    const float bz0 = bias[c0];
    const float bz1 = bias[c1];
#pragma unroll
    for (int r = 0; r < 8; ++r) {
        const int row0 = m0 + r + h * 8;
        const int row1 = row0 + 16;
        C[(size_t)row0 * ldc + c0] = acc00[r] + bz0;
        C[(size_t)row0 * ldc + c1] = acc01[r] + bz1;
        C[(size_t)row1 * ldc + c0] = acc10[r] + bz0;
        C[(size_t)row1 * ldc + c1] = acc11[r] + bz1;
    }
}

// ---------------------------------------------------------------------------
// Depthwise causal conv (k=4) + bias + SiLU.  Reads xc half of xz (fp32),
// writes bf16 activations ready for GEMM2.
// ---------------------------------------------------------------------------
__global__ void conv_silu_kernel(const float* __restrict__ xz,
                                 const float* __restrict__ conv_w,
                                 const float* __restrict__ conv_b,
                                 bf16_t* __restrict__ xcbf) {
    int idx = blockIdx.x * blockDim.x + threadIdx.x;
    if (idx >= ROWS * DINNER) return;
    const int ch = idx % DINNER;
    const int t  = (idx / DINNER) % SEQ;
    const int b  = idx / (DINNER * SEQ);

    const float* w = conv_w + ch * DCONV;
    float acc = conv_b[ch];
#pragma unroll
    for (int k = 0; k < DCONV; ++k) {
        const int ts = t - (DCONV - 1) + k;
        if (ts >= 0)
            acc += w[k] * xz[(size_t)(b * SEQ + ts) * NXZ + ch];
    }
    const float s = acc / (1.0f + __expf(-acc));      // SiLU
    xcbf[(size_t)(b * SEQ + t) * DINNER + ch] = f2bf(s);
}

// ---------------------------------------------------------------------------
// Selective scan.  One thread per (batch, channel); 16 fp32 states in
// registers.  Per timestep the 32 shared values (d_raw[16], B[16]) are staged
// in LDS by the first wave of the block.  Fuses: delta = softplus(d_raw@W_dt
// + b_dt), dA = exp(delta*A), recurrence, y = (sum(s)+u)*Dp, y *= silu(z),
// emits bf16 for GEMM3.
// ---------------------------------------------------------------------------
__global__ __launch_bounds__(256, 1)
void scan_kernel(const float* __restrict__ dbu,   // ROWS x 2080
                 const float* __restrict__ xz,    // ROWS x 4096 (z at col 2048+)
                 const float* __restrict__ A,     // DINNER x 16  (= -exp(A_log))
                 const float* __restrict__ Wdt,   // DINNER x 16
                 const float* __restrict__ bdt,   // DINNER
                 const float* __restrict__ Dp,    // DINNER
                 bf16_t* __restrict__ ybf) {      // ROWS x DINNER
    const int blocksPerBatch = DINNER / 256;       // 8
    const int b  = blockIdx.x / blocksPerBatch;
    const int ch = (blockIdx.x % blocksPerBatch) * 256 + threadIdx.x;

    __shared__ float sDB[2 * DSTATE];              // d_raw[0..15], B[16..31]

    float Ar[DSTATE], Wr[DSTATE];
#pragma unroll
    for (int j = 0; j < DSTATE; ++j) {
        Ar[j] = A[ch * DSTATE + j];
        Wr[j] = Wdt[ch * DSTATE + j];
    }
    const float bdtc = bdt[ch];
    const float Dpc  = Dp[ch];

    float s[DSTATE];
#pragma unroll
    for (int j = 0; j < DSTATE; ++j) s[j] = 0.0f;

    for (int t = 0; t < SEQ; ++t) {
        const size_t row = (size_t)b * SEQ + t;
        if (threadIdx.x < 2 * DSTATE)
            sDB[threadIdx.x] = dbu[row * NDBU + threadIdx.x];
        __syncthreads();

        // delta = softplus(d_raw @ W_dt[ch] + b_dt[ch])
        float dt = bdtc;
#pragma unroll
        for (int j = 0; j < DSTATE; ++j) dt += sDB[j] * Wr[j];
        const float delta = (dt > 20.0f) ? dt : log1pf(__expf(dt));

        const float u  = dbu[row * NDBU + 2 * DSTATE + ch];
        const float du = delta * u;

        float y = 0.0f;
#pragma unroll
        for (int j = 0; j < DSTATE; ++j) {
            const float dA = __expf(delta * Ar[j]);
            s[j] = dA * s[j] + du * sDB[DSTATE + j];
            y += s[j];
        }
        y = (y + u) * Dpc;

        const float z  = xz[row * NXZ + DINNER + ch];
        const float yf = y * (z / (1.0f + __expf(-z)));
        ybf[row * DINNER + ch] = f2bf(yf);
        __syncthreads();
    }
}

// ---------------------------------------------------------------------------
// Workspace layout (bytes)
// ---------------------------------------------------------------------------
#define OFF_XBF     ((size_t)0)                                   //  4 MB
#define OFF_WINBF   (OFF_XBF   + (size_t)ROWS   * DMODEL * 2)     //  8 MB
#define OFF_WXBF    (OFF_WINBF + (size_t)NXZ    * DMODEL * 2)     //  8.5 MB
#define OFF_WOUTBF  (OFF_WXBF  + (size_t)NDBU   * DINNER * 2)     //  4 MB
#define OFF_A       (OFF_WOUTBF+ (size_t)DMODEL * DINNER * 2)     //  128 KB
#define OFF_XZ      (OFF_A     + (size_t)DINNER * DSTATE * 4)     // 32 MB
#define OFF_XCBF    (OFF_XZ    + (size_t)ROWS   * NXZ    * 4)     //  8 MB
#define OFF_DBU     (OFF_XCBF  + (size_t)ROWS   * DINNER * 2)     // 17 MB
#define OFF_YBF     (OFF_DBU   + (size_t)ROWS   * NDBU   * 4)     //  8 MB

extern "C" void kernel_launch(void* const* d_in, const int* in_sizes, int n_in,
                              void* d_out, int out_size, void* d_ws, size_t ws_size,
                              hipStream_t stream) {
    const float* x      = (const float*)d_in[0];
    const float* W_in   = (const float*)d_in[1];
    const float* b_in   = (const float*)d_in[2];
    const float* conv_w = (const float*)d_in[3];
    const float* conv_b = (const float*)d_in[4];
    const float* W_x    = (const float*)d_in[5];
    const float* b_x    = (const float*)d_in[6];
    const float* W_dt   = (const float*)d_in[7];
    const float* b_dt   = (const float*)d_in[8];
    const float* A_log  = (const float*)d_in[9];
    const float* Dp     = (const float*)d_in[10];
    const float* W_out  = (const float*)d_in[11];
    const float* b_out  = (const float*)d_in[12];
    float* out = (float*)d_out;

    char* ws = (char*)d_ws;
    bf16_t* xbf    = (bf16_t*)(ws + OFF_XBF);
    bf16_t* Winbf  = (bf16_t*)(ws + OFF_WINBF);
    bf16_t* Wxbf   = (bf16_t*)(ws + OFF_WXBF);
    bf16_t* Woutbf = (bf16_t*)(ws + OFF_WOUTBF);
    float*  Aneg   = (float*) (ws + OFF_A);
    float*  xz     = (float*) (ws + OFF_XZ);
    bf16_t* xcbf   = (bf16_t*)(ws + OFF_XCBF);
    float*  dbu    = (float*) (ws + OFF_DBU);
    bf16_t* ybf    = (bf16_t*)(ws + OFF_YBF);

    const int T = 256;
    // --- precision conversion (once per launch) ---
    {
        int n;
        n = ROWS * DMODEL;
        cvt_f32_to_bf16<<<(n + T - 1) / T, T, 0, stream>>>(x, xbf, n);
        n = NXZ * DMODEL;
        cvt_f32_to_bf16<<<(n + T - 1) / T, T, 0, stream>>>(W_in, Winbf, n);
        n = NDBU * DINNER;
        cvt_f32_to_bf16<<<(n + T - 1) / T, T, 0, stream>>>(W_x, Wxbf, n);
        n = DMODEL * DINNER;
        cvt_f32_to_bf16<<<(n + T - 1) / T, T, 0, stream>>>(W_out, Woutbf, n);
        n = DINNER * DSTATE;
        neg_exp_kernel<<<(n + T - 1) / T, T, 0, stream>>>(A_log, Aneg, n);
    }

    // --- GEMM1: xz = x @ W_in.T + b_in   (2048 x 4096, K=1024) ---
    {
        dim3 grid((NXZ + 63) / 64, ROWS / 128);
        gemm_bf16_wmma<<<grid, 256, 0, stream>>>(xbf, Winbf, b_in, xz,
                                                 ROWS, NXZ, DMODEL, NXZ);
    }

    // --- depthwise causal conv + SiLU -> bf16 ---
    {
        int n = ROWS * DINNER;
        conv_silu_kernel<<<(n + T - 1) / T, T, 0, stream>>>(xz, conv_w, conv_b, xcbf);
    }

    // --- GEMM2: dbu = xc @ W_x.T + b_x   (2048 x 2080, K=2048) ---
    {
        dim3 grid((NDBU + 63) / 64, ROWS / 128);
        gemm_bf16_wmma<<<grid, 256, 0, stream>>>(xcbf, Wxbf, b_x, dbu,
                                                 ROWS, NDBU, DINNER, NDBU);
    }

    // --- selective scan (fused delta/softplus/recurrence/gate) ---
    {
        dim3 grid(BATCH * (DINNER / 256));
        scan_kernel<<<grid, 256, 0, stream>>>(dbu, xz, Aneg, W_dt, b_dt, Dp, ybf);
    }

    // --- GEMM3: out = y @ W_out.T + b_out  (2048 x 1024, K=2048) ---
    {
        dim3 grid((DMODEL + 63) / 64, ROWS / 128);
        gemm_bf16_wmma<<<grid, 256, 0, stream>>>(ybf, Woutbf, b_out, out,
                                                 ROWS, DMODEL, DINNER, DMODEL);
    }
}
